// PseudotimeModel_37074157699316
// MI455X (gfx1250) — matrix operand order repaired and checked by reading, not compiled, not run on hardware
//
#include <hip/hip_runtime.h>
#include <hip/hip_bf16.h>
#include <stdint.h>

// ---------------------------------------------------------------------------
// Types for CDNA5 WMMA (wave32): 16x16x32 bf16 -> f32 accumulate
// ---------------------------------------------------------------------------
typedef __attribute__((ext_vector_type(16))) __bf16 v16bf;
typedef __attribute__((ext_vector_type(8)))  __bf16 v8bf;
typedef __attribute__((ext_vector_type(8)))  float  v8f;

#define TM 128     // block tile rows
#define TN 64      // block tile cols
#define LDA 40     // LDS row stride in bf16 elems (80B: 16B-aligned, bank-staggered)

#if defined(__gfx1250__)
#define USE_ASYNC_LDS 1
#endif

// ---------------------------------------------------------------------------
// Small utility kernels
// ---------------------------------------------------------------------------
__global__ void k_fill_f32(float* p, float v, int n) {
  int i = blockIdx.x * blockDim.x + threadIdx.x;
  if (i < n) p[i] = v;
}

__global__ void k_degree(const int* __restrict__ dst, float* __restrict__ deg, int E) {
  int e = blockIdx.x * blockDim.x + threadIdx.x;
  if (e < E) unsafeAtomicAdd(&deg[dst[e]], 1.0f);
}

__global__ void k_rsqrt(float* p, int n) {
  int i = blockIdx.x * blockDim.x + threadIdx.x;
  if (i < n) p[i] = rsqrtf(p[i]);
}

// W [K, C] row-major (f32) -> Wt [C, K] bf16 (transposed so B fragments are
// 16 contiguous K-values per lane, the same K order the A fragments use)
__global__ void k_cvt_w(const float* __restrict__ W, __bf16* __restrict__ Wt,
                        int K, int C) {
  int idx = blockIdx.x * blockDim.x + threadIdx.x;
  if (idx >= K * C) return;
  int k = idx / C;
  int c = idx - k * C;
  Wt[(size_t)c * K + k] = (__bf16)W[idx];
}

// Row-gathered f32 -> bf16 convert (rowmap==nullptr for identity; perm for DGI
// corruption). F is a power of two (1024).
__global__ void k_cvt_rows(const float* __restrict__ X, const int* __restrict__ rowmap,
                           __bf16* __restrict__ Xb, int n, int F, int fbits) {
  size_t idx = (size_t)blockIdx.x * blockDim.x + threadIdx.x;
  if (idx >= ((size_t)n << fbits)) return;
  int i = (int)(idx >> fbits);
  int f = (int)(idx & (F - 1));
  int r = rowmap ? rowmap[i] : i;
  Xb[idx] = (__bf16)X[(size_t)r * F + f];
}

// ---------------------------------------------------------------------------
// Async 16B copy global -> LDS (CDNA5 GLOBAL_LOAD_ASYNC_TO_LDS_B128, tracked
// by ASYNCcnt). Inline asm per CDNA5 ISA: VDST = LDS
// address VGPR, VADDR = 64-bit global address VGPR pair.
// ---------------------------------------------------------------------------
__device__ __forceinline__ void async_cp16(const __bf16* g, __bf16* l) {
#ifdef USE_ASYNC_LDS
  uint32_t laddr = (uint32_t)(uintptr_t)l;     // LDS byte offset (low 32 bits)
  uint64_t gaddr = (uint64_t)(uintptr_t)g;
  asm volatile("global_load_async_to_lds_b128 %0, %1, off"
               :: "v"(laddr), "v"(gaddr)
               : "memory");
#endif
}

__device__ __forceinline__ void wait_async0() {
#ifdef USE_ASYNC_LDS
  asm volatile("s_wait_asynccnt 0x0" ::: "memory");
#endif
}

// ---------------------------------------------------------------------------
// WMMA GEMM: C[M,NC] (f32) = A[M,K] (bf16, row-major) x Bt[NC,K] (bf16, B
// stored transposed). Block: 256 threads = 8 waves -> 128x64 tile; each wave
// owns a 32x32 sub-tile = 2x2 WMMA 16x16 accumulators.
// A tile is staged in LDS via GLOBAL_LOAD_ASYNC_TO_LDS_B128 with double
// buffering (ASYNCcnt), overlapping the next K-slab fetch with the WMMAs.
// ---------------------------------------------------------------------------
__global__ __launch_bounds__(256)
void k_gemm_bf16(const __bf16* __restrict__ A, const __bf16* __restrict__ Bt,
                 float* __restrict__ C, int M, int K, int NC) {
#ifdef USE_ASYNC_LDS
  __shared__ __bf16 sA[2 * TM * LDA];
#else
  __shared__ __bf16 sA[TM * LDA];
#endif

  const int tid  = threadIdx.x;
  const int lane = tid & 31;
  const int wid  = tid >> 5;
  const int wm   = wid >> 1;   // 0..3 : 32-row band
  const int wn   = wid & 1;    // 0..1 : 32-col band
  const int lr   = lane & 15;  // row/col within 16x16 fragment
  const int lh   = lane >> 4;  // half-wave select (K interleave)

  const int rowBlk = blockIdx.x * TM;
  const int colBlk = blockIdx.y * TN;

  v8f acc[2][2] = {};

  // A-tile cooperative load mapping: 256 threads x 16 bf16 = 128x32 tile
  const int ldr = tid >> 1;          // 0..127 tile row
  const int ldk = (tid & 1) * 16;    // 0 or 16 within 32-wide K slab
  int gr = rowBlk + ldr;
  if (gr >= M) gr = M - 1;           // clamp (stores are guarded)
  const __bf16* aRow = A + (size_t)gr * K;

  const int nsteps = K >> 5;

#ifdef USE_ASYNC_LDS
  // prologue: slab 0 -> buffer 0
  {
    __bf16* l0 = sA + ldr * LDA + ldk;
    async_cp16(aRow + ldk,     l0);
    async_cp16(aRow + ldk + 8, l0 + 8);
    wait_async0();
    __syncthreads();
  }
#endif

  for (int step = 0; step < nsteps; ++step) {
    const int k0 = step << 5;
#ifdef USE_ASYNC_LDS
    const __bf16* base = sA + (step & 1) * (TM * LDA);
    if (step + 1 < nsteps) {           // prefetch next slab into other buffer
      const __bf16* gsrc = aRow + k0 + 32 + ldk;
      __bf16* ldst = sA + ((step + 1) & 1) * (TM * LDA) + ldr * LDA + ldk;
      async_cp16(gsrc,     ldst);
      async_cp16(gsrc + 8, ldst + 8);
    }
#else
    const __bf16* base = sA;
    __syncthreads();
    {
      const v8bf* s = (const v8bf*)(aRow + k0 + ldk);
      *(v8bf*)(sA + ldr * LDA + ldk)     = s[0];
      *(v8bf*)(sA + ldr * LDA + ldk + 8) = s[1];
    }
    __syncthreads();
#endif

    // A fragments: lane lr holds row M=lr; elements 0..15 = K(lh*16 + 0..15)
    // (contiguous, matching B fragment K order)
    v16bf af[2];
#pragma unroll
    for (int mt = 0; mt < 2; ++mt) {
      const __bf16* ap = base + (wm * 32 + mt * 16 + lr) * LDA + lh * 16;
      v8bf lo = *(const v8bf*)(ap);
      v8bf hi = *(const v8bf*)(ap + 8);
      af[mt] = __builtin_shufflevector(lo, hi,
               0, 1, 2, 3, 4, 5, 6, 7, 8, 9, 10, 11, 12, 13, 14, 15);
    }

    // B fragments: lane lr = column N; elements 0..15 = K(lh*16 + 0..15),
    // contiguous in Bt because B is stored transposed.
    v16bf bfr[2];
#pragma unroll
    for (int nt = 0; nt < 2; ++nt) {
      int col = colBlk + wn * 32 + nt * 16 + lr;
      const __bf16* bp = Bt + (size_t)col * K + k0 + lh * 16;
      v8bf lo = *(const v8bf*)(bp);
      v8bf hi = *(const v8bf*)(bp + 8);
      bfr[nt] = __builtin_shufflevector(lo, hi,
                0, 1, 2, 3, 4, 5, 6, 7, 8, 9, 10, 11, 12, 13, 14, 15);
    }

#pragma unroll
    for (int mt = 0; mt < 2; ++mt)
#pragma unroll
      for (int nt = 0; nt < 2; ++nt)
        acc[mt][nt] = __builtin_amdgcn_wmma_f32_16x16x32_bf16(
            false, af[mt], false, bfr[nt], (short)0, acc[mt][nt], false, false);

#ifdef USE_ASYNC_LDS
    wait_async0();       // my slab copy landed
    __syncthreads();     // everyone's landed; everyone done reading cur buffer
#endif
  }

  // D layout: VGPR g -> row g (+8 for upper half-wave), lane lr = column.
  // Fast unguarded path for full row-blocks (only the last block is partial).
  const bool full = (rowBlk + TM) <= M;
#pragma unroll
  for (int mt = 0; mt < 2; ++mt)
#pragma unroll
    for (int nt = 0; nt < 2; ++nt) {
      int rBase = rowBlk + wm * 32 + mt * 16 + lh * 8;
      int c     = colBlk + wn * 32 + nt * 16 + lr;
      float* cp = C + (size_t)rBase * NC + c;
      if (full) {
#pragma unroll
        for (int g = 0; g < 8; ++g) { *cp = acc[mt][nt][g]; cp += NC; }
      } else {
#pragma unroll
        for (int g = 0; g < 8; ++g) {
          if (rBase + g < M) *cp = acc[mt][nt][g];
          cp += NC;
        }
      }
    }
}

// ---------------------------------------------------------------------------
// Edge gather/scale/scatter-add: agg[d] += h[s] * dinv[s]*dinv[d]
// One thread per (edge, feature); 32 consecutive lanes share an edge (H=512)
// so h loads and atomics are coalesced.
// ---------------------------------------------------------------------------
__global__ void k_edge_agg(const int* __restrict__ src, const int* __restrict__ dst,
                           const float* __restrict__ dinv, const float* __restrict__ h,
                           float* __restrict__ agg, int E, int H, int hbits) {
  size_t idx = (size_t)blockIdx.x * blockDim.x + threadIdx.x;
  if (idx >= ((size_t)E << hbits)) return;
  int e = (int)(idx >> hbits);
  int j = (int)(idx & (H - 1));
  int s = src[e];
  int d = dst[e];
  float nrm = dinv[s] * dinv[d];
  float v = h[(size_t)s * H + j] * nrm;
  unsafeAtomicAdd(&agg[(size_t)d * H + j], v);
}

// Fused self-loop + bias + PReLU -> bf16 (feeds next WMMA GEMM)
__global__ void k_finalize_bf16(const float* __restrict__ agg, const float* __restrict__ lin,
                                const float* __restrict__ dinv, const float* __restrict__ b,
                                const float* __restrict__ a, __bf16* __restrict__ out,
                                int n, int H, int hbits) {
  size_t idx = (size_t)blockIdx.x * blockDim.x + threadIdx.x;
  if (idx >= ((size_t)n << hbits)) return;
  int i = (int)(idx >> hbits);
  int j = (int)(idx & (H - 1));
  float di = dinv[i];
  float v = agg[idx] + lin[idx] * di * di + b[j];
  v = (v >= 0.0f) ? v : a[j] * v;
  out[idx] = (__bf16)v;
}

// Fused self-loop + bias + PReLU -> f32 (final layer output z / neg_z)
__global__ void k_finalize_f32(const float* __restrict__ agg, const float* __restrict__ lin,
                               const float* __restrict__ dinv, const float* __restrict__ b,
                               const float* __restrict__ a, float* __restrict__ out,
                               int n, int H, int hbits) {
  size_t idx = (size_t)blockIdx.x * blockDim.x + threadIdx.x;
  if (idx >= ((size_t)n << hbits)) return;
  int i = (int)(idx >> hbits);
  int j = (int)(idx & (H - 1));
  float di = dinv[i];
  float v = agg[idx] + lin[idx] * di * di + b[j];
  v = (v >= 0.0f) ? v : a[j] * v;
  out[idx] = v;
}

// Column partial sums: blockDim.x == H; thread j accumulates rows with
// coalesced access, one atomic per (block, column).
__global__ void k_colsum(const float* __restrict__ z, float* __restrict__ sumbuf,
                         int n, int H) {
  int j = threadIdx.x;
  float s = 0.0f;
  for (int r = blockIdx.x; r < n; r += gridDim.x)
    s += z[(size_t)r * H + j];
  unsafeAtomicAdd(&sumbuf[j], s);
}

__global__ void k_sigmoid(const float* __restrict__ sumbuf, float* __restrict__ out,
                          float invn, int H) {
  int j = threadIdx.x;
  if (j < H) out[j] = 1.0f / (1.0f + __expf(-sumbuf[j] * invn));
}

// ---------------------------------------------------------------------------
// Host side
// ---------------------------------------------------------------------------
extern "C" void kernel_launch(void* const* d_in, const int* in_sizes, int n_in,
                              void* d_out, int out_size, void* d_ws, size_t ws_size,
                              hipStream_t stream) {
  const float* x    = (const float*)d_in[0];
  const int*   ei   = (const int*)d_in[1];
  const int*   perm = (const int*)d_in[2];
  const float* W1   = (const float*)d_in[3];
  const float* b1   = (const float*)d_in[4];
  const float* a1   = (const float*)d_in[5];
  const float* W2   = (const float*)d_in[6];
  const float* b2   = (const float*)d_in[7];
  const float* a2   = (const float*)d_in[8];

  const int E = in_sizes[1] / 2;
  const int N = in_sizes[2];
  const int H = in_sizes[4];
  const int F = in_sizes[0] / N;
  const int hbits = 31 - __builtin_clz((unsigned)H);   // H = 512
  const int fbits = 31 - __builtin_clz((unsigned)F);   // F = 1024

  const int* srcp = ei;
  const int* dstp = ei + E;

  float* out  = (float*)d_out;
  float* z    = out;
  float* negz = out + (size_t)N * H;
  float* summ = out + 2 * (size_t)N * H;

  // ---- workspace carve-out (256B-aligned) ----
  char* ws = (char*)d_ws;
  auto carve = [&](size_t bytes) -> char* {
    char* p = ws;
    ws += (bytes + 255) & ~(size_t)255;
    return p;
  };
  float*  dinv   = (float*)carve((size_t)N * sizeof(float));
  float*  sumbuf = (float*)carve((size_t)H * sizeof(float));
  __bf16* W1t    = (__bf16*)carve((size_t)F * H * sizeof(__bf16));
  __bf16* W2t    = (__bf16*)carve((size_t)H * H * sizeof(__bf16));
  __bf16* xbf    = (__bf16*)carve((size_t)N * F * sizeof(__bf16));
  __bf16* hbf    = (__bf16*)carve((size_t)N * H * sizeof(__bf16));
  float*  tmp    = (float*)carve((size_t)N * H * sizeof(float));
  float*  agg    = (float*)carve((size_t)N * H * sizeof(float));

  const int TPB = 256;
  const size_t nh = (size_t)N << hbits;
  const size_t eh = (size_t)E << hbits;
  const size_t nf = (size_t)N << fbits;

  // ---- degree / normalization (shared by both encoders) ----
  k_fill_f32<<<(N + TPB - 1) / TPB, TPB, 0, stream>>>(dinv, 1.0f, N);  // self-loop
  k_degree<<<(E + TPB - 1) / TPB, TPB, 0, stream>>>(dstp, dinv, E);
  k_rsqrt<<<(N + TPB - 1) / TPB, TPB, 0, stream>>>(dinv, N);

  // ---- weights -> bf16 transposed ----
  k_cvt_w<<<(F * H + TPB - 1) / TPB, TPB, 0, stream>>>(W1, W1t, F, H);
  k_cvt_w<<<(H * H + TPB - 1) / TPB, TPB, 0, stream>>>(W2, W2t, H, H);

  dim3 ggrid((N + TM - 1) / TM, H / TN);

  auto run_encoder = [&](const int* rowmap, float* zout) {
    // layer 1: h = PReLU(Agg(x @ W1) + b1)
    k_cvt_rows<<<(unsigned)((nf + TPB - 1) / TPB), TPB, 0, stream>>>(
        x, rowmap, xbf, N, F, fbits);
    k_gemm_bf16<<<ggrid, TPB, 0, stream>>>(xbf, W1t, tmp, N, F, H);
    (void)hipMemsetAsync(agg, 0, nh * sizeof(float), stream);
    k_edge_agg<<<(unsigned)((eh + TPB - 1) / TPB), TPB, 0, stream>>>(
        srcp, dstp, dinv, tmp, agg, E, H, hbits);
    k_finalize_bf16<<<(unsigned)((nh + TPB - 1) / TPB), TPB, 0, stream>>>(
        agg, tmp, dinv, b1, a1, hbf, N, H, hbits);

    // layer 2: z = PReLU(Agg(h @ W2) + b2)
    k_gemm_bf16<<<ggrid, TPB, 0, stream>>>(hbf, W2t, tmp, N, H, H);
    (void)hipMemsetAsync(agg, 0, nh * sizeof(float), stream);
    k_edge_agg<<<(unsigned)((eh + TPB - 1) / TPB), TPB, 0, stream>>>(
        srcp, dstp, dinv, tmp, agg, E, H, hbits);
    k_finalize_f32<<<(unsigned)((nh + TPB - 1) / TPB), TPB, 0, stream>>>(
        agg, tmp, dinv, b2, a2, zout, N, H, hbits);
  };

  run_encoder(nullptr, z);   // positive samples
  run_encoder(perm, negz);   // DGI corruption: shuffled node features

  // ---- summary = sigmoid(mean(z, axis=0)) ----
  (void)hipMemsetAsync(sumbuf, 0, (size_t)H * sizeof(float), stream);
  k_colsum<<<512, H, 0, stream>>>(z, sumbuf, N, H);
  k_sigmoid<<<1, H, 0, stream>>>(sumbuf, summ, 1.0f / (float)N, H);
}
